// MultiHeadAttention_80479097193000
// MI455X (gfx1250) — matrix operand order repaired and checked by reading, not compiled, not run on hardware
//
#include <hip/hip_runtime.h>
#include <hip/hip_bf16.h>

typedef __bf16 bf16;
typedef __attribute__((ext_vector_type(16))) __bf16 v16bf;
typedef __attribute__((ext_vector_type(8)))  __bf16 v8bf;
typedef __attribute__((ext_vector_type(8)))  float   v8f;
typedef __attribute__((ext_vector_type(4)))  float   v4f;

#define D_MODEL 1024
#define T_SEQ   2048
#define BATCH   4
#define NHEAD   16
#define HD      64
#define ROWS    (BATCH * T_SEQ)   // 8192

static __device__ __forceinline__ v8f wmma_bf16(v16bf a, v16bf b, v8f c) {
  // D = A(16x32) * B(32x16) + C, f32 accumulate
  return __builtin_amdgcn_wmma_f32_16x16x32_bf16(false, a, false, b, (short)0, c,
                                                 false, false);
}

// --------------------------------------------------------------------------
// Weight transpose + f32 -> bf16:  Wt[n][k] = (bf16)W[k][n]
// --------------------------------------------------------------------------
__global__ __launch_bounds__(256) void transpose_w(const float* __restrict__ W,
                                                   bf16* __restrict__ Wt) {
  __shared__ float tile[32][33];
  const int bx = blockIdx.x & 31;   // n tile
  const int by = blockIdx.x >> 5;   // k tile
  const int tx = threadIdx.x & 31;
  const int ty = threadIdx.x >> 5;  // 0..7
#pragma unroll
  for (int i = 0; i < 32; i += 8)
    tile[ty + i][tx] = W[(size_t)(by * 32 + ty + i) * D_MODEL + bx * 32 + tx];
  __syncthreads();
#pragma unroll
  for (int i = 0; i < 32; i += 8)
    Wt[(size_t)(bx * 32 + ty + i) * D_MODEL + by * 32 + tx] = (bf16)tile[tx][ty + i];
}

// --------------------------------------------------------------------------
// bf16 WMMA GEMM:  C[8192 x 1024] = A[8192 x 1024] * Wt^T + bias
//   Wt stored transposed [N][K] so B-operand loads are contiguous.
//   One wave computes a 32(M) x 64(N) tile, looping K in steps of 32.
// MODE 0: out bf16, head-split   [B,H,T,64]   (Q, K projections)
// MODE 1: out bf16, head-split-T [B,H,64,T]   (V projection, transposed)
// MODE 2: out f32, row-major     [8192,1024]  (output projection -> d_out)
// --------------------------------------------------------------------------
template <bool A_IS_F32, int MODE>
__global__ __launch_bounds__(256) void gemm_wmma(const void* __restrict__ Av,
                                                 const bf16* __restrict__ Wt,
                                                 const float* __restrict__ bias,
                                                 void* __restrict__ Cv) {
  const int lane   = threadIdx.x & 31;
  const int wid    = blockIdx.x * 8 + (threadIdx.x >> 5);
  const int m0     = (wid & 255) * 32;     // 256 M-tiles of 32
  const int n0     = (wid >> 8) * 64;      // 16  N-tiles of 64
  const int nl     = lane & 15;
  const int kA     = (lane & 16) >> 1;     // A half: element offset 0 / 8
  const int kB     = (lane & 16) >> 1;     // B half: element offset 0 / 8
  const int subrow = (lane & 16) >> 1;     // accumulator row offset 0 / 8

  const v8f vzero = {0.f, 0.f, 0.f, 0.f, 0.f, 0.f, 0.f, 0.f};
  v8f acc[2][4];
#pragma unroll
  for (int i = 0; i < 2; ++i)
#pragma unroll
    for (int j = 0; j < 4; ++j) acc[i][j] = vzero;

  const float* Af = (const float*)Av;
  const bf16*  Ab = (const bf16*)Av;

  for (int kb = 0; kb < D_MODEL; kb += 32) {
    v16bf a0, a1;
    if (A_IS_F32) {
      const float* r0 = Af + (size_t)(m0 + nl) * D_MODEL + kb + kA;
      const float* r1 = r0 + 16 * D_MODEL;
      __builtin_prefetch(r0 + 32, 0, 0);   // next K block -> global_prefetch
      v4f x0 = *(const v4f*)(r0 + 0);
      v4f x1 = *(const v4f*)(r0 + 4);
      v4f x2 = *(const v4f*)(r0 + 16);
      v4f x3 = *(const v4f*)(r0 + 20);
#pragma unroll
      for (int i = 0; i < 4; ++i) {
        a0[i] = (bf16)x0[i]; a0[4 + i] = (bf16)x1[i];
        a0[8 + i] = (bf16)x2[i]; a0[12 + i] = (bf16)x3[i];
      }
      x0 = *(const v4f*)(r1 + 0);
      x1 = *(const v4f*)(r1 + 4);
      x2 = *(const v4f*)(r1 + 16);
      x3 = *(const v4f*)(r1 + 20);
#pragma unroll
      for (int i = 0; i < 4; ++i) {
        a1[i] = (bf16)x0[i]; a1[4 + i] = (bf16)x1[i];
        a1[8 + i] = (bf16)x2[i]; a1[12 + i] = (bf16)x3[i];
      }
    } else {
      const bf16* r0 = Ab + (size_t)(m0 + nl) * D_MODEL + kb + kA;
      const bf16* r1 = r0 + 16 * D_MODEL;
      *(v8bf*)&a0       = *(const v8bf*)(r0);
      *((v8bf*)&a0 + 1) = *(const v8bf*)(r0 + 16);
      *(v8bf*)&a1       = *(const v8bf*)(r1);
      *((v8bf*)&a1 + 1) = *(const v8bf*)(r1 + 16);
    }
#pragma unroll
    for (int j = 0; j < 4; ++j) {
      const bf16* rb = Wt + (size_t)(n0 + j * 16 + nl) * D_MODEL + kb + kB;
      v16bf b;
      *(v8bf*)&b       = *(const v8bf*)(rb);
      *((v8bf*)&b + 1) = *(const v8bf*)(rb + 16);
      acc[0][j] = wmma_bf16(a0, b, acc[0][j]);
      acc[1][j] = wmma_bf16(a1, b, acc[1][j]);
    }
  }

  // Epilogue: element r -> row (r + subrow), column (lane & 15)
#pragma unroll
  for (int i = 0; i < 2; ++i) {
#pragma unroll
    for (int j = 0; j < 4; ++j) {
#pragma unroll
      for (int r = 0; r < 8; ++r) {
        const int row = m0 + i * 16 + r + subrow;
        const int col = n0 + j * 16 + nl;
        const float val = acc[i][j][r] + bias[col];
        if (MODE == 2) {
          ((float*)Cv)[(size_t)row * D_MODEL + col] = val;
        } else {
          const int bb = row >> 11, t = row & (T_SEQ - 1);
          const int h = col >> 6, d = col & (HD - 1);
          const bf16 vb = (bf16)val;
          if (MODE == 0)
            ((bf16*)Cv)[(((size_t)(bb * NHEAD + h) * T_SEQ) + t) * HD + d] = vb;
          else
            ((bf16*)Cv)[(((size_t)(bb * NHEAD + h) * HD) + d) * T_SEQ + t] = vb;
        }
      }
    }
  }
}

// --------------------------------------------------------------------------
// Flash attention: one wave owns 16 queries of one (b,h); loops 32-key blocks.
// Qh,Kh: [B,H,T,64] bf16; Vt: [B,H,64,T] bf16; O: [B,T,D] bf16 (head-interleaved)
// --------------------------------------------------------------------------
__global__ __launch_bounds__(256) void attention_fa(const bf16* __restrict__ Qh,
                                                    const bf16* __restrict__ Kh,
                                                    const bf16* __restrict__ Vt,
                                                    bf16* __restrict__ O) {
  __shared__ __align__(16) bf16 plds[8 * 16 * 32];  // 1 KB per wave
  const int lane   = threadIdx.x & 31;
  const int wv     = threadIdx.x >> 5;
  const int job    = blockIdx.x * 8 + wv;
  const int qt     = job & (T_SEQ / 16 - 1);  // query tile 0..127
  const int bh     = job >> 7;                // (b*16 + h) 0..63
  const int nl     = lane & 15;
  const int kA     = (lane & 16) >> 1;
  const int kB     = (lane & 16) >> 1;
  const int subrow = (lane & 16) >> 1;

  const bf16* Q = Qh + (size_t)bh * T_SEQ * HD + (size_t)qt * 16 * HD;
  const bf16* K = Kh + (size_t)bh * T_SEQ * HD;
  const bf16* V = Vt + (size_t)bh * HD * T_SEQ;

  // Q A-operand tiles: dims 0..31 and 32..63
  v16bf qa0, qa1;
  {
    const bf16* r = Q + nl * HD + kA;
    *(v8bf*)&qa0       = *(const v8bf*)(r);
    *((v8bf*)&qa0 + 1) = *(const v8bf*)(r + 16);
    *(v8bf*)&qa1       = *(const v8bf*)(r + 32);
    *((v8bf*)&qa1 + 1) = *(const v8bf*)(r + 48);
  }

  const v8f vzero = {0.f, 0.f, 0.f, 0.f, 0.f, 0.f, 0.f, 0.f};
  v8f o0 = vzero, o1 = vzero, o2 = vzero, o3 = vzero;
  float mrun[8], lrun[8];
#pragma unroll
  for (int r = 0; r < 8; ++r) { mrun[r] = -1e30f; lrun[r] = 0.f; }

  bf16* myp = plds + wv * 512;

  for (int kb = 0; kb < T_SEQ; kb += 32) {
    // K^T B-operand tiles: (keys kb..kb+15 | kb+16..kb+31) x (dims 0..31 | 32..63)
    const bf16* k0 = K + (size_t)(kb + nl) * HD + kB;
    const bf16* k1 = k0 + 16 * HD;
    v16bf b00, b01, b10, b11;
    *(v8bf*)&b00       = *(const v8bf*)(k0);
    *((v8bf*)&b00 + 1) = *(const v8bf*)(k0 + 16);
    *(v8bf*)&b01       = *(const v8bf*)(k0 + 32);
    *((v8bf*)&b01 + 1) = *(const v8bf*)(k0 + 48);
    *(v8bf*)&b10       = *(const v8bf*)(k1);
    *((v8bf*)&b10 + 1) = *(const v8bf*)(k1 + 16);
    *(v8bf*)&b11       = *(const v8bf*)(k1 + 32);
    *((v8bf*)&b11 + 1) = *(const v8bf*)(k1 + 48);

    v8f s0 = vzero, s1 = vzero;
    s0 = wmma_bf16(qa0, b00, s0);
    s0 = wmma_bf16(qa1, b01, s0);
    s1 = wmma_bf16(qa0, b10, s1);
    s1 = wmma_bf16(qa1, b11, s1);

    // ---- online softmax over the 32 new columns (f32) ----
    float pm[8];
#pragma unroll
    for (int r = 0; r < 8; ++r) {
      const float a = s0[r] * 0.125f;  // 1/sqrt(64)
      const float b = s1[r] * 0.125f;
      s0[r] = a; s1[r] = b;
      pm[r] = fmaxf(a, b);
    }
#pragma unroll
    for (int off = 1; off < 16; off <<= 1)
#pragma unroll
      for (int r = 0; r < 8; ++r)
        pm[r] = fmaxf(pm[r], __shfl_xor(pm[r], off, 32));

    float fs[8], rs[8];
#pragma unroll
    for (int r = 0; r < 8; ++r) {
      const float mn = fmaxf(mrun[r], pm[r]);
      fs[r] = __expf(mrun[r] - mn);
      const float p0 = __expf(s0[r] - mn);
      const float p1 = __expf(s1[r] - mn);
      mrun[r] = mn;
      s0[r] = p0; s1[r] = p1;
      rs[r] = p0 + p1;
    }
#pragma unroll
    for (int off = 1; off < 16; off <<= 1)
#pragma unroll
      for (int r = 0; r < 8; ++r)
        rs[r] += __shfl_xor(rs[r], off, 32);
#pragma unroll
    for (int r = 0; r < 8; ++r) {
      lrun[r] = lrun[r] * fs[r] + rs[r];
      o0[r] *= fs[r]; o1[r] *= fs[r]; o2[r] *= fs[r]; o3[r] *= fs[r];
    }

    // ---- P (16x32) -> LDS (row-major bf16), reload in A-operand layout ----
#pragma unroll
    for (int r = 0; r < 8; ++r) {
      const int row = r + subrow;
      myp[row * 32 + nl]      = (bf16)s0[r];
      myp[row * 32 + 16 + nl] = (bf16)s1[r];
    }
    asm volatile("s_wait_dscnt 0x0" ::: "memory");
    v16bf pa;
    {
      const bf16* r = myp + nl * 32 + kA;
      *(v8bf*)&pa       = *(const v8bf*)(r);
      *((v8bf*)&pa + 1) = *(const v8bf*)(r + 16);
    }

    // ---- O += P @ V  (Vt gives contiguous B-operand rows) ----
#pragma unroll
    for (int j = 0; j < 4; ++j) {
      const bf16* vr = V + (size_t)(j * 16 + nl) * T_SEQ + kb + kB;
      v16bf vb;
      *(v8bf*)&vb       = *(const v8bf*)(vr);
      *((v8bf*)&vb + 1) = *(const v8bf*)(vr + 16);
      if (j == 0) o0 = wmma_bf16(pa, vb, o0);
      if (j == 1) o1 = wmma_bf16(pa, vb, o1);
      if (j == 2) o2 = wmma_bf16(pa, vb, o2);
      if (j == 3) o3 = wmma_bf16(pa, vb, o3);
    }
  }

  // ---- normalize and scatter to [B,T,D] bf16 (head-interleaved) ----
  const int bb = bh >> 4, h = bh & 15;
#pragma unroll
  for (int r = 0; r < 8; ++r) {
    const float inv = 1.0f / lrun[r];
    const int t = qt * 16 + r + subrow;
    const size_t base = ((size_t)(bb * T_SEQ + t)) * D_MODEL + h * HD;
    O[base + 0  + nl] = (bf16)(o0[r] * inv);
    O[base + 16 + nl] = (bf16)(o1[r] * inv);
    O[base + 32 + nl] = (bf16)(o2[r] * inv);
    O[base + 48 + nl] = (bf16)(o3[r] * inv);
  }
}

// --------------------------------------------------------------------------
extern "C" void kernel_launch(void* const* d_in, const int* in_sizes, int n_in,
                              void* d_out, int out_size, void* d_ws, size_t ws_size,
                              hipStream_t stream) {
  (void)in_sizes; (void)n_in; (void)out_size; (void)ws_size;
  const float* q   = (const float*)d_in[0];
  const float* k   = (const float*)d_in[1];
  const float* v   = (const float*)d_in[2];
  const float* w_q = (const float*)d_in[3];
  const float* b_q = (const float*)d_in[4];
  const float* w_k = (const float*)d_in[5];
  const float* b_k = (const float*)d_in[6];
  const float* w_v = (const float*)d_in[7];
  const float* b_v = (const float*)d_in[8];
  const float* w_o = (const float*)d_in[9];
  const float* b_o = (const float*)d_in[10];
  float* out = (float*)d_out;

  const size_t WT_ELEMS = (size_t)D_MODEL * D_MODEL;          // 1M
  const size_t H_ELEMS  = (size_t)BATCH * NHEAD * T_SEQ * HD; // 8M
  bf16* wtq = (bf16*)d_ws;
  bf16* wtk = wtq + WT_ELEMS;
  bf16* wtv = wtk + WT_ELEMS;
  bf16* wto = wtv + WT_ELEMS;
  bf16* qh  = wto + WT_ELEMS;
  bf16* kh  = qh + H_ELEMS;
  bf16* vt  = kh + H_ELEMS;
  bf16* ao  = vt + H_ELEMS;   // total 72 MB of workspace

  transpose_w<<<1024, 256, 0, stream>>>(w_q, wtq);
  transpose_w<<<1024, 256, 0, stream>>>(w_k, wtk);
  transpose_w<<<1024, 256, 0, stream>>>(w_v, wtv);
  transpose_w<<<1024, 256, 0, stream>>>(w_o, wto);

  // 4096 wave tiles (32x64) / 8 waves per block = 512 blocks
  gemm_wmma<true, 0><<<512, 256, 0, stream>>>((const void*)q, wtq, b_q, (void*)qh);
  gemm_wmma<true, 0><<<512, 256, 0, stream>>>((const void*)k, wtk, b_k, (void*)kh);
  gemm_wmma<true, 1><<<512, 256, 0, stream>>>((const void*)v, wtv, b_v, (void*)vt);

  // 64 (b,h) * 128 query tiles = 8192 wave jobs / 8 = 1024 blocks
  attention_fa<<<1024, 256, 0, stream>>>(qh, kh, vt, ao);

  gemm_wmma<false, 2><<<512, 256, 0, stream>>>((const void*)ao, wto, b_o, (void*)out);
}
